// SRUModel_28166395527468
// MI455X (gfx1250) — matrix operand order repaired and checked by reading, not compiled, not run on hardware
//
#include <hip/hip_runtime.h>
#include <hip/hip_bf16.h>

// ---------------------------------------------------------------------------
// SRU language model forward for MI455X (gfx1250, wave32, WMMA + TDM).
// Pipeline: embed -> (bf16 GEMM -> SRU scan) x2 -> LayerNorm -> bf16 GEMM+bias
// GEMM A-tiles are staged into LDS by the Tensor Data Mover (tensor_load_to_lds
// with LDS padding producing the exact stride the WMMA fragment gather wants).
// ---------------------------------------------------------------------------

#define VOCAB 32000
#define DMODEL 1024
#define NLAYER 2
#define BATCH 8
#define SEQ 512
#define MROWS (BATCH * SEQ)   // 4096

typedef __attribute__((ext_vector_type(16))) __bf16 v16bf;
typedef __attribute__((ext_vector_type(8)))  __bf16 v8bf;
typedef __attribute__((ext_vector_type(8)))  float  v8f;
typedef __attribute__((ext_vector_type(4)))  unsigned int v4u;
typedef __attribute__((ext_vector_type(8)))  int v8i;
typedef __attribute__((ext_vector_type(4)))  int v4i;

__device__ __forceinline__ unsigned short f32_to_bf16_rne(float f) {
    unsigned int u = __float_as_uint(f);
    unsigned int r = u + 0x7FFFu + ((u >> 16) & 1u);   // round-to-nearest-even
    return (unsigned short)(r >> 16);
}

// ---------------------------------------------------------------------------
// fp32 -> bf16 bulk convert (grid-strided)
// ---------------------------------------------------------------------------
__global__ void cvt_f32_bf16_kernel(const float* __restrict__ src,
                                    unsigned short* __restrict__ dst, int n) {
    int i = blockIdx.x * blockDim.x + threadIdx.x;
    int stride = gridDim.x * blockDim.x;
    for (; i < n; i += stride) dst[i] = f32_to_bf16_rne(src[i]);
}

// ---------------------------------------------------------------------------
// Embedding gather: h[(t*B+b)*D + d] = emb[x[b*T+t]*D + d]  (time-major h)
// ---------------------------------------------------------------------------
__global__ void embed_kernel(const int* __restrict__ x,
                             const float* __restrict__ emb,
                             float* __restrict__ h) {
    int bt = blockIdx.x;                 // b*T + t
    int b = bt >> 9;                     // / SEQ
    int t = bt & (SEQ - 1);
    int tok = x[bt];
    const float* src = emb + (size_t)tok * DMODEL;
    float* dst = h + (size_t)(t * BATCH + b) * DMODEL;
    for (int d = threadIdx.x; d < DMODEL; d += blockDim.x) dst[d] = src[d];
}

// ---------------------------------------------------------------------------
// bf16 WMMA GEMM:  C[M x N] (+bias) = A[M x K] * Bm[K x N], row-major.
// 256 threads = 8 waves (2 x 4).  Block tile 64 x 256, wave tile 32 x 64,
// K-step 32 (one v_wmma_f32_16x16x32_bf16 per 16x16 subtile per step).
//
// A tile (64 x 32 bf16) is DMA'd by the Tensor Data Mover: 2D descriptor,
// data_size=2B, tile_dim0=32, tile_dim1=64, tensor_dim0_stride=K, with LDS
// padding (pad_interval=3 -> every 16 DWORDs, pad_amount=3 -> 4 DWORDs) so
// each 64B row lands on an 80B (40-half) LDS stride -- exactly the layout the
// WMMA A-fragment gather below expects (K = e(+16) + 8*(lane>=16)).
// B is staged transposed ([n][k]) by all threads so the B fragment
// (K = e(+16) + 8*(lane>=16)) is two ds_load_b128 per lane.
// ---------------------------------------------------------------------------
__global__ __launch_bounds__(256)
void gemm_bf16_wmma(const unsigned short* __restrict__ A,
                    const unsigned short* __restrict__ Bm,
                    float* __restrict__ C,
                    const float* __restrict__ bias,
                    int N, int K) {
    const int n0 = blockIdx.x * 256;
    const int m0 = blockIdx.y * 64;

    const int tid   = threadIdx.x;
    const int lane  = tid & 31;
    const int wave  = tid >> 5;
    const int wm    = wave >> 2;            // 0..1
    const int wn    = wave & 3;             // 0..3
    const int lhalf = (lane >= 16) ? 1 : 0;
    const int lmod  = lane & 15;

    __shared__ __align__(16) unsigned short As[64 * 40];    // [row][k], stride 40
    __shared__ __align__(16) unsigned short Bs[256 * 40];   // [col][k], stride 40

    v8f acc[2][4];
#pragma unroll
    for (int mi = 0; mi < 2; ++mi)
#pragma unroll
        for (int ni = 0; ni < 4; ++ni) acc[mi][ni] = 0.0f;

    // TDM descriptor constants for the A tile (group0 global addr updated per step)
    const unsigned lds_a = (unsigned)(uintptr_t)(&As[0]);   // low 32 bits = LDS offset
    // group1: data_size=2B (bits17:16=1), pad_enable (bit20),
    //         pad_interval=3 (bits24:22, 16 DWORDs), pad_amount=3 (bits31:25, 4 DWORDs)
    const v8i g1 = { (int)0x06D10000u,
                     (int)(((unsigned)K & 0xFFFFu) << 16),   // tensor_dim0[15:0] (=K)
                     (int)0x10000000u,                       // tensor_dim1[15:0]=4096
                     (int)(32u << 16),                       // tile_dim0=32 (dim1 hi = 0)
                     64,                                     // tile_dim1=64 rows
                     K,                                      // tensor_dim0_stride lo
                     0, 0 };
    const v4i g2 = { 0, 0, 0, 0 };
    const v4i g3 = { 0, 0, 0, 0 };

    for (int k0 = 0; k0 < K; k0 += 32) {
        // ---- kick off TDM for the A tile (one issue; EXEC ignored by TDM) ----
        if (wave == 0) {
            unsigned long long ga =
                (unsigned long long)(uintptr_t)(A + (size_t)m0 * K + k0);
            v4u g0 = { 1u,                                   // count=1, user D#
                       lds_a,
                       (unsigned)ga,
                       ((unsigned)(ga >> 32) & 0x01FFFFFFu) | 0x80000000u }; // type=2
#if __clang_major__ >= 23
            v8i gz = { 0, 0, 0, 0, 0, 0, 0, 0 };
            __builtin_amdgcn_tensor_load_to_lds(g0, g1, g2, g3, gz, 0);
#else
            __builtin_amdgcn_tensor_load_to_lds(g0, g1, g2, g3, 0);
#endif
        }

        // ---- stage B tile: 32 x 256 bf16, transposed into Bs[col][k] ----
        {
            int col  = (tid & 31) << 3;
            int krow = tid >> 5;
#pragma unroll
            for (int p = 0; p < 4; ++p) {
                int k = p * 8 + krow;
                const unsigned short* gp = &Bm[(size_t)(k0 + k) * N + n0 + col];
                if (k0 + 32 < K)
                    __builtin_prefetch(gp + (size_t)32 * N, 0, 3);  // global_prefetch_b8
                uint4 v = *reinterpret_cast<const uint4*>(gp);
                const unsigned short* ph = reinterpret_cast<const unsigned short*>(&v);
#pragma unroll
                for (int j = 0; j < 8; ++j) Bs[(col + j) * 40 + k] = ph[j];
            }
        }

        // ---- TDM completion (issuing wave), then block-wide visibility ----
        if (wave == 0) __builtin_amdgcn_s_wait_tensorcnt(0);
        __syncthreads();

        // ---- gather fragments ----
        v16bf afrag[2];
#pragma unroll
        for (int mi = 0; mi < 2; ++mi) {
            int row  = wm * 32 + mi * 16 + lmod;
            int base = row * 40 + lhalf * 8;
            v8bf lo = *reinterpret_cast<const v8bf*>(&As[base]);
            v8bf hi = *reinterpret_cast<const v8bf*>(&As[base + 16]);
            afrag[mi] = __builtin_shufflevector(lo, hi,
                0, 1, 2, 3, 4, 5, 6, 7, 8, 9, 10, 11, 12, 13, 14, 15);
        }
        v16bf bfrag[4];
#pragma unroll
        for (int ni = 0; ni < 4; ++ni) {
            int col  = wn * 64 + ni * 16 + lmod;
            int base = col * 40 + lhalf * 8;
            v8bf lo = *reinterpret_cast<const v8bf*>(&Bs[base]);
            v8bf hi = *reinterpret_cast<const v8bf*>(&Bs[base + 16]);
            bfrag[ni] = __builtin_shufflevector(lo, hi,
                0, 1, 2, 3, 4, 5, 6, 7, 8, 9, 10, 11, 12, 13, 14, 15);
        }

        // ---- 8 WMMAs per wave per K-step ----
#pragma unroll
        for (int mi = 0; mi < 2; ++mi)
#pragma unroll
            for (int ni = 0; ni < 4; ++ni)
                acc[mi][ni] = __builtin_amdgcn_wmma_f32_16x16x32_bf16(
                    false, afrag[mi], false, bfrag[ni],
                    (short)0, acc[mi][ni], false, false);
        __syncthreads();
    }

    // ---- epilogue: C/D layout -> VGPR j holds M = j + 8*(lane>=16) ----
#pragma unroll
    for (int mi = 0; mi < 2; ++mi) {
        int rbase = m0 + wm * 32 + mi * 16 + lhalf * 8;
#pragma unroll
        for (int ni = 0; ni < 4; ++ni) {
            int col = n0 + wn * 64 + ni * 16 + lmod;
            float bv = bias ? bias[col] : 0.0f;
#pragma unroll
            for (int j = 0; j < 8; ++j)
                C[(size_t)(rbase + j) * N + col] = acc[mi][ni][j] + bv;
        }
    }
}

// ---------------------------------------------------------------------------
// SRU elementwise recurrence (sequential in T, parallel over B*D channels).
// U rows are time-major (row = t*B + b), split [xt | fi | ri] along 3D.
// ---------------------------------------------------------------------------
__global__ void sru_scan_kernel(const float* __restrict__ U,
                                float* __restrict__ h,
                                const float* __restrict__ bvec) {
    int idx = blockIdx.x * blockDim.x + threadIdx.x;   // 0 .. B*D-1
    int b = idx >> 10;                                 // / DMODEL
    int d = idx & (DMODEL - 1);
    float bf = bvec[d];
    float br = bvec[DMODEL + d];
    float c = 0.0f;
    for (int t = 0; t < SEQ; ++t) {
        size_t row = (size_t)(t * BATCH + b);
        const float* u = U + row * (3 * DMODEL);
        float xt = u[d];
        float fi = u[DMODEL + d];
        float ri = u[2 * DMODEL + d];
        float f = 1.0f / (1.0f + __expf(-(fi + bf)));
        float r = 1.0f / (1.0f + __expf(-(ri + br)));
        float x = h[row * DMODEL + d];
        c = f * c + (1.0f - f) * xt;
        h[row * DMODEL + d] = r * tanhf(c) + (1.0f - r) * x;
    }
}

// ---------------------------------------------------------------------------
// LayerNorm over D, reading time-major h and writing bf16 rows reordered to
// batch-major (row j = b*T + t) as input to the output projection.
// ---------------------------------------------------------------------------
__global__ void layernorm_kernel(const float* __restrict__ h,
                                 const float* __restrict__ g,
                                 const float* __restrict__ be,
                                 unsigned short* __restrict__ hb) {
    int j = blockIdx.x;                 // b*T + t
    int b = j >> 9;
    int t = j & (SEQ - 1);
    const float* row = h + (size_t)(t * BATCH + b) * DMODEL;

    float vals[4];
    float sum = 0.0f, sq = 0.0f;
#pragma unroll
    for (int i = 0; i < 4; ++i) {
        float v = row[threadIdx.x + i * 256];
        vals[i] = v; sum += v; sq += v * v;
    }
    __shared__ float s1[256], s2[256];
    s1[threadIdx.x] = sum; s2[threadIdx.x] = sq;
    __syncthreads();
    for (int s = 128; s > 0; s >>= 1) {
        if (threadIdx.x < s) {
            s1[threadIdx.x] += s1[threadIdx.x + s];
            s2[threadIdx.x] += s2[threadIdx.x + s];
        }
        __syncthreads();
    }
    float mu  = s1[0] * (1.0f / DMODEL);
    float var = s2[0] * (1.0f / DMODEL) - mu * mu;
    float inv = rsqrtf(var + 1e-5f);

    unsigned short* dst = hb + (size_t)j * DMODEL;
#pragma unroll
    for (int i = 0; i < 4; ++i) {
        int d = threadIdx.x + i * 256;
        float v = (vals[i] - mu) * inv * g[d] + be[d];
        dst[d] = f32_to_bf16_rne(v);
    }
}

// ---------------------------------------------------------------------------
// Launch
// ---------------------------------------------------------------------------
extern "C" void kernel_launch(void* const* d_in, const int* in_sizes, int n_in,
                              void* d_out, int out_size, void* d_ws, size_t ws_size,
                              hipStream_t stream) {
    const int*   x     = (const int*)d_in[0];
    const float* emb   = (const float*)d_in[1];
    const float* W     = (const float*)d_in[2];
    const float* bvec  = (const float*)d_in[3];
    const float* ln_g  = (const float*)d_in[4];
    const float* ln_b  = (const float*)d_in[5];
    const float* out_W = (const float*)d_in[6];
    const float* out_b = (const float*)d_in[7];
    float* out = (float*)d_out;

    // workspace carve (all offsets 256B-aligned)
    char* ws = (char*)d_ws;
    float*          h   = (float*)ws;          ws += (size_t)MROWS * DMODEL * 4;      // 16 MB
    unsigned short* hb  = (unsigned short*)ws; ws += (size_t)MROWS * DMODEL * 2;      // 8 MB
    float*          U   = (float*)ws;          ws += (size_t)MROWS * 3 * DMODEL * 4;  // 48 MB
    unsigned short* Wb  = (unsigned short*)ws; ws += (size_t)NLAYER * DMODEL * 3 * DMODEL * 2; // 12 MB
    unsigned short* oWb = (unsigned short*)ws;                                        // 64 MB

    // one-time (per launch) weight conversion to bf16
    cvt_f32_bf16_kernel<<<2048, 256, 0, stream>>>(W, Wb, NLAYER * DMODEL * 3 * DMODEL);
    cvt_f32_bf16_kernel<<<4096, 256, 0, stream>>>(out_W, oWb, DMODEL * VOCAB);

    // embedding gather -> time-major h
    embed_kernel<<<MROWS, 256, 0, stream>>>(x, emb, h);

    // SRU layers
    for (int l = 0; l < NLAYER; ++l) {
        cvt_f32_bf16_kernel<<<2048, 256, 0, stream>>>(h, hb, MROWS * DMODEL);
        gemm_bf16_wmma<<<dim3(3 * DMODEL / 256, MROWS / 64), 256, 0, stream>>>(
            hb, Wb + (size_t)l * DMODEL * 3 * DMODEL, U, nullptr, 3 * DMODEL, DMODEL);
        sru_scan_kernel<<<(BATCH * DMODEL) / 256, 256, 0, stream>>>(
            U, h, bvec + l * 2 * DMODEL);
    }

    // LayerNorm (reorders to batch-major bf16)
    layernorm_kernel<<<MROWS, 256, 0, stream>>>(h, ln_g, ln_b, hb);

    // output projection: [4096 x 1024] x [1024 x 32000] + out_b -> d_out
    gemm_bf16_wmma<<<dim3(VOCAB / 256, MROWS / 64), 256, 0, stream>>>(
        hb, oWb, out, out_b, VOCAB, DMODEL);
}
